// MaskMamba_16106127360161
// MI455X (gfx1250) — matrix operand order repaired and checked by reading, not compiled, not run on hardware
//
#include <hip/hip_runtime.h>
#include <cstdint>
#include <cstddef>

// ---------------- types for WMMA ----------------
typedef __bf16 bf16_t;
typedef __attribute__((ext_vector_type(16))) __bf16 v16bf;
typedef __attribute__((ext_vector_type(8)))  float  v8f;

// ---------------- problem constants ----------------
#define NBATCH  8
#define GRP     256
#define NPT     32
#define FEATC   273
#define ENCD    384
#define DMODEL  384
#define DINNER  768
#define LSEQ    103           // visible tokens per batch (256 - int(0.6*256))
#define TOK     (NBATCH*LSEQ) // 824
#define TOKP    832           // padded to 52*16
#define DEPTHL  12
#define DTR     24
#define DSTATE  16
#define GTOT    (NBATCH*GRP)  // 2048
#define GCH     256           // encoder groups per chunk
#define MCH     (GCH*NPT)     // 8192 rows per chunk (512 M-tiles)
#define KENC    576           // 546 padded to 18*32
#define NEN1    576           // conv1 out-ch padded to 36*16 (mult of 64)

// ---------------- small device helpers ----------------
__device__ __forceinline__ unsigned f2key(float x) {
  unsigned u = __float_as_uint(x);
  return (u & 0x80000000u) ? ~u : (u | 0x80000000u);
}
__device__ __forceinline__ float key2f(unsigned k) {
  unsigned u = (k & 0x80000000u) ? (k ^ 0x80000000u) : ~k;
  return __uint_as_float(u);
}
__device__ __forceinline__ float siluf(float x) { return x / (1.f + expf(-x)); }
__device__ __forceinline__ float softplusf(float x) {
  return (x > 20.f) ? x : log1pf(expf(x));
}
__device__ __forceinline__ float gelu_exact(float x) {
  return 0.5f * x * (1.f + erff(x * 0.70710678118654752f));
}

// =========================================================
// WMMA GEMM:  C[M x N] = A[M x K] * Bt[N x K]^T   (bf16 in, f32 acc)
// One wave computes a 16x64 tile (1 M-tile x 4 N-tiles); the A
// fragment is loaded once per K-step and reused by 4 WMMAs.
// Software-pipelined: fragments for step ks+1 are issued before the
// WMMAs of step ks so global-load latency overlaps matrix math.
// All pointers kept as distinct scalars so address-space inference
// keeps every fragment load on the GLOBAL path (LOADcnt only, no
// FLAT/DScnt coupling).
// Nt4 = Ntiles/4 (all N dims padded to multiples of 64).
// MODE 0: store f32 C (ldc = stride)
// MODE 1: y = relu(c*sc[n]+sh[n]) -> bf16 Obf (ldc = stride)
// MODE 2: order-preserving-key atomicMax into keys[(gb+m/32)*ENCD+n]
// =========================================================
template <int MODE>
__global__ void gemm_wmma(const bf16_t* __restrict__ A,
                          const bf16_t* __restrict__ Bt,
                          float* __restrict__ C,
                          bf16_t* __restrict__ Obf,
                          const float* __restrict__ sc,
                          const float* __restrict__ sh,
                          unsigned* __restrict__ keys, int group_base,
                          int Mtiles, int Nt4, int Ksteps,
                          int lda, int ldb, int ldc) {
  int wave = (int)((blockIdx.x * blockDim.x + threadIdx.x) >> 5);
  if (wave >= Mtiles * Nt4) return;        // whole-wave uniform exit: EXEC stays all-ones
  int lane = (int)(threadIdx.x & 31);
  int tm = wave / Nt4, tn0 = (wave % Nt4) * 4;
  int half = lane >> 4;
  int l15  = lane & 15;

  const bf16_t* pA  = A + (size_t)(tm * 16 + l15) * lda;
  const bf16_t* pB0 = Bt + (size_t)((tn0 + 0) * 16 + l15) * ldb;
  const bf16_t* pB1 = Bt + (size_t)((tn0 + 1) * 16 + l15) * ldb;
  const bf16_t* pB2 = Bt + (size_t)((tn0 + 2) * 16 + l15) * ldb;
  const bf16_t* pB3 = Bt + (size_t)((tn0 + 3) * 16 + l15) * ldb;

  union Frag { v16bf v; uint4 q[2]; };
  // A frag (16x32 bf16, ISA 7.12.2): lane m=l15; two 16B slabs per lane
  auto ldA = [&](const bf16_t* p, Frag& f) {
    f.q[0] = *(const uint4*)(p + half * 8);
    f.q[1] = *(const uint4*)(p + 16 + half * 8);
  };
  // B frag (32x16 bf16): lane n=l15; same K order as A (two 16B slabs per lane)
  auto ldB = [&](const bf16_t* p, Frag& f) {
    f.q[0] = *(const uint4*)(p + half * 8);
    f.q[1] = *(const uint4*)(p + 16 + half * 8);
  };

  v8f acc0 = {}, acc1 = {}, acc2 = {}, acc3 = {};
  Frag a, b0, b1, b2, b3;
  ldA(pA, a);
  ldB(pB0, b0); ldB(pB1, b1); ldB(pB2, b2); ldB(pB3, b3);

  for (int ks = 0; ks < Ksteps - 1; ++ks) {
    __builtin_prefetch(pA + 64, 0, 3);     // near-temporal prefetch 2 steps ahead
    __builtin_prefetch(pB0 + 64, 0, 3);
    __builtin_prefetch(pB2 + 64, 0, 3);
    Frag an, bn0, bn1, bn2, bn3;
    ldA(pA + 32, an);                      // issue next-step loads first
    ldB(pB0 + 32, bn0); ldB(pB1 + 32, bn1);
    ldB(pB2 + 32, bn2); ldB(pB3 + 32, bn3);
    // math on current fragments
    acc0 = __builtin_amdgcn_wmma_f32_16x16x32_bf16(false, a.v, false, b0.v, (short)0, acc0, false, false);
    acc1 = __builtin_amdgcn_wmma_f32_16x16x32_bf16(false, a.v, false, b1.v, (short)0, acc1, false, false);
    acc2 = __builtin_amdgcn_wmma_f32_16x16x32_bf16(false, a.v, false, b2.v, (short)0, acc2, false, false);
    acc3 = __builtin_amdgcn_wmma_f32_16x16x32_bf16(false, a.v, false, b3.v, (short)0, acc3, false, false);
    a = an; b0 = bn0; b1 = bn1; b2 = bn2; b3 = bn3;
    pA += 32; pB0 += 32; pB1 += 32; pB2 += 32; pB3 += 32;
  }
  // drain final step
  acc0 = __builtin_amdgcn_wmma_f32_16x16x32_bf16(false, a.v, false, b0.v, (short)0, acc0, false, false);
  acc1 = __builtin_amdgcn_wmma_f32_16x16x32_bf16(false, a.v, false, b1.v, (short)0, acc1, false, false);
  acc2 = __builtin_amdgcn_wmma_f32_16x16x32_bf16(false, a.v, false, b2.v, (short)0, acc2, false, false);
  acc3 = __builtin_amdgcn_wmma_f32_16x16x32_bf16(false, a.v, false, b3.v, (short)0, acc3, false, false);

  auto epilogue = [&](int j, const v8f& acc) {
    int n = (tn0 + j) * 16 + l15;
#pragma unroll
    for (int r = 0; r < 8; ++r) {
      int m = tm * 16 + half * 8 + r;      // C layout: lanes 0-15 rows 0-7, 16-31 rows 8-15
      float val = acc[r];
      if (MODE == 0) {
        C[(size_t)m * ldc + n] = val;
      } else if (MODE == 1) {
        float y = val * sc[n] + sh[n];
        Obf[(size_t)m * ldc + n] = (bf16_t)(y > 0.f ? y : 0.f);
      } else {
        int grp = group_base + (m >> 5);
        atomicMax(&keys[(size_t)grp * ENCD + n], f2key(val));
      }
    }
  };
  epilogue(0, acc0); epilogue(1, acc1); epilogue(2, acc2); epilogue(3, acc3);
}

// ---------------- fills / converts ----------------
__global__ void fill_u32(unsigned* p, unsigned v, long n) {
  long i = (long)blockIdx.x * blockDim.x + threadIdx.x;
  if (i < n) p[i] = v;
}
__global__ void fill_bf16z(bf16_t* p, long n) {
  long i = (long)blockIdx.x * blockDim.x + threadIdx.x;
  if (i < n) p[i] = (bf16_t)0.f;
}
// dst[l][rp][cp] = (rp<R && cp<C) ? src[l][rp][cp] : 0   (f32 -> bf16, pad)
__global__ void pad_convert(bf16_t* dst, const float* src,
                            int L, int R, int C, int Rp, int Cp) {
  long i = (long)blockIdx.x * blockDim.x + threadIdx.x;
  long total = (long)L * Rp * Cp;
  if (i >= total) return;
  int cp = (int)(i % Cp);
  long t = i / Cp;
  int rp = (int)(t % Rp);
  int l  = (int)(t / Rp);
  float v = (rp < R && cp < C) ? src[((long)l * R + rp) * C + cp] : 0.f;
  dst[i] = (bf16_t)v;
}

// ---------------- encoder stages ----------------
// fused BN scale/shift: y = (Wf + c1b - bn_m)*s + bn_b,  s = g*rsqrt(v+eps)
__global__ void bn_prep(float* s, float* t, const float* g, const float* b,
                        const float* m, const float* v, const float* c1b) {
  int c = blockIdx.x * blockDim.x + threadIdx.x;
  if (c >= NEN1) return;
  if (c < 546) {
    float sc = g[c] * rsqrtf(v[c] + 1e-5f);
    s[c] = sc;
    t[c] = (c1b[c] - m[c]) * sc + b[c];
  } else { s[c] = 0.f; t[c] = 0.f; }
}
// per-group per-feature max over the 32 points
__global__ void group_max(float* gmax, const float* nb) {
  long i = (long)blockIdx.x * blockDim.x + threadIdx.x;
  if (i >= (long)GTOT * FEATC) return;
  int c = (int)(i % FEATC);
  long g = i / FEATC;
  float m = nb[(g * NPT) * FEATC + c];
  for (int n = 1; n < NPT; ++n)
    m = fmaxf(m, nb[(g * NPT + n) * FEATC + c]);
  gmax[i] = m;
}
// F[(g_local*32+n)][c] : c<273 -> group max (broadcast), 273..545 -> point feat, pad 0
__global__ void build_F(bf16_t* F, const float* gmax, const float* nb, int gbase) {
  long i = (long)blockIdx.x * blockDim.x + threadIdx.x;
  if (i >= (long)MCH * KENC) return;
  int c = (int)(i % KENC);
  long r = i / KENC;
  int n  = (int)(r & 31);
  long g = gbase + (r >> 5);
  float v = 0.f;
  if (c < FEATC)          v = gmax[g * FEATC + c];
  else if (c < 2 * FEATC) v = nb[(g * NPT + n) * FEATC + (c - FEATC)];
  F[i] = (bf16_t)v;
}
__global__ void finish_tokens(float* tok, const unsigned* keys, const float* c2b) {
  long i = (long)blockIdx.x * blockDim.x + threadIdx.x;
  if (i >= (long)GTOT * ENCD) return;
  tok[i] = key2f(keys[i]) + c2b[i % ENCD];
}

// ---------------- masking / pos-embed ----------------
__global__ void vis_build(int* vis, const unsigned char* mask) {
  int b = threadIdx.x;
  if (b >= NBATCH) return;
  int cnt = 0;
  for (int g = 0; g < GRP; ++g)
    if (!mask[b * GRP + g]) { if (cnt < LSEQ) vis[b * LSEQ + cnt] = g; ++cnt; }
}
__global__ void pos_mid(float* mid, const float* center, const int* vis,
                        const float* w1, const float* b1) {
  long i = (long)blockIdx.x * blockDim.x + threadIdx.x;
  if (i >= (long)TOK * 128) return;
  int m = (int)(i % 128); int t = (int)(i / 128);
  int b = t / LSEQ, j = t % LSEQ;
  int g = vis[b * LSEQ + j];
  const float* mc = center + ((long)b * GRP + g) * 3;
  float v = w1[m * 3] * mc[0] + w1[m * 3 + 1] * mc[1] + w1[m * 3 + 2] * mc[2] + b1[m];
  mid[i] = gelu_exact(v);
}
__global__ void pos_h0(float* h, const float* tok, const float* mid, const int* vis,
                       const float* w2, const float* b2) {
  long i = (long)blockIdx.x * blockDim.x + threadIdx.x;
  if (i >= (long)TOK * DMODEL) return;
  int d = (int)(i % DMODEL); int t = (int)(i / DMODEL);
  int b = t / LSEQ, j = t % LSEQ;
  int g = vis[b * LSEQ + j];
  float acc = b2[d];
  const float* mrow = mid + (long)t * 128;
  const float* wrow = w2 + (long)d * 128;
  for (int k = 0; k < 128; ++k) acc += mrow[k] * wrow[k];
  h[(long)t * DMODEL + d] = tok[((long)b * GRP + g) * ENCD + d] + acc;
}

// ---------------- residual + LayerNorm -> bf16 hn ----------------
__device__ __forceinline__ void blk_reduce2(float& sum, float& sq,
                                            float* s1, float* s2, int tid) {
  s1[tid] = sum; s2[tid] = sq; __syncthreads();
  for (int off = 64; off; off >>= 1) {
    if (tid < off) { s1[tid] += s1[tid + off]; s2[tid] += s2[tid + off]; }
    __syncthreads();
  }
  sum = s1[0]; sq = s2[0]; __syncthreads();
}
__global__ void resln_kernel(const float* h, float* res, bf16_t* hn,
                             const float* lnw, const float* lnb, int first) {
  __shared__ float s1[128], s2[128];
  int t = blockIdx.x, tid = threadIdx.x;
  float v[3]; float sum = 0.f, sq = 0.f;
#pragma unroll
  for (int i = 0; i < 3; ++i) {
    int d = tid + i * 128;
    float x = h[(long)t * DMODEL + d];
    if (!first) x += res[(long)t * DMODEL + d];
    res[(long)t * DMODEL + d] = x;
    v[i] = x; sum += x; sq += x * x;
  }
  blk_reduce2(sum, sq, s1, s2, tid);
  float mu = sum / DMODEL;
  float rs = rsqrtf(sq / DMODEL - mu * mu + 1e-5f);
#pragma unroll
  for (int i = 0; i < 3; ++i) {
    int d = tid + i * 128;
    hn[(long)t * DMODEL + d] = (bf16_t)((v[i] - mu) * rs * lnw[d] + lnb[d]);
  }
}

// ---------------- depthwise causal conv (k=4) + SiLU ----------------
__global__ void conv_silu(float* xc, bf16_t* xcb, const float* xz,
                          const float* cw, const float* cb) {
  long i = (long)blockIdx.x * blockDim.x + threadIdx.x;
  if (i >= (long)TOK * DINNER) return;
  int c = (int)(i % DINNER); int t = (int)(i / DINNER);
  int b = t / LSEQ, l = t % LSEQ;
  float acc = cb[c];
#pragma unroll
  for (int j = 0; j < 4; ++j) {
    int li = l - 3 + j;
    if (li >= 0) acc += cw[c * 4 + j] * xz[((long)(b * LSEQ + li)) * (2 * DINNER) + c];
  }
  float y = siluf(acc);
  xc[i]  = y;
  xcb[i] = (bf16_t)y;
}

// dtA[r][c] = (r<TOK && c<24) ? dbl[r][c] : 0   (bf16, K padded 24->32)
__global__ void dt_copy(bf16_t* dtA, const float* dbl) {
  long i = (long)blockIdx.x * blockDim.x + threadIdx.x;
  if (i >= (long)TOKP * 32) return;
  int c = (int)(i % 32); int r = (int)(i / 32);
  float v = (r < TOK && c < DTR) ? dbl[(long)r * 64 + c] : 0.f;
  dtA[i] = (bf16_t)v;
}

// ---------------- selective scan ----------------
// Block = 256 channels of one batch (grid 8*3). The shared B/C panel
// (103 steps x 32) is staged global->LDS with gfx1250 async-to-LDS
// ops (ASYNCcnt), then every lane runs its 16-state recurrence.
__global__ void scan_kernel(const float* dtl, const float* dpb_l, const float* xc,
                            const float* dbl, const float* alog_l, const float* dpar_l,
                            const float* xz, bf16_t* ymul) {
  __shared__ float bc[LSEQ * 32];        // [l][s]: s<16 -> B, s>=16 -> C
  int b = blockIdx.x / 3;
  int c = (blockIdx.x % 3) * 256 + threadIdx.x;

  // async stage of the (b, :, 24..55) slice of dbl into LDS
  for (int idx = threadIdx.x; idx < LSEQ * 32; idx += 256) {
    int l = idx >> 5, s = idx & 31;
    unsigned lds_off = (unsigned)(uintptr_t)(&bc[idx]);  // low 32 bits = LDS offset
    unsigned long long ga =
        (unsigned long long)(uintptr_t)(dbl + (size_t)(b * LSEQ + l) * 64 + DTR + s);
    asm volatile("global_load_async_to_lds_b32 %0, %1, off"
                 :: "v"(lds_off), "v"(ga) : "memory");
  }
  asm volatile("s_wait_asynccnt 0" ::: "memory");
  __syncthreads();

  float a[DSTATE], hst[DSTATE];
#pragma unroll
  for (int s = 0; s < DSTATE; ++s) { a[s] = -expf(alog_l[c * DSTATE + s]); hst[s] = 0.f; }
  float dpb = dpb_l[c], Dp = dpar_l[c];
  for (int l = 0; l < LSEQ; ++l) {
    int t = b * LSEQ + l;
    float dtv = softplusf(dtl[(long)t * DINNER + c] + dpb);
    float xv  = xc[(long)t * DINNER + c];
    float dx  = dtv * xv;
    float y = 0.f;
    const float* dr = &bc[l * 32];
#pragma unroll
    for (int s = 0; s < DSTATE; ++s) {
      float dA = expf(dtv * a[s]);
      hst[s] = dA * hst[s] + dx * dr[s];             // B
      y += hst[s] * dr[DSTATE + s];                  // C
    }
    y += xv * Dp;
    float zv = xz[(long)t * (2 * DINNER) + DINNER + c];
    ymul[(long)t * DINNER + c] = (bf16_t)(y * siluf(zv));
  }
}

// ---------------- final: res+=h, LN(normf), LN(norm2) ----------------
__global__ void final_ln_kernel(float* out, const float* h, const float* res,
                                const float* nfw, const float* nfb,
                                const float* n2w, const float* n2b) {
  __shared__ float s1[128], s2[128];
  int t = blockIdx.x, tid = threadIdx.x;
  float v[3]; float sum = 0.f, sq = 0.f;
#pragma unroll
  for (int i = 0; i < 3; ++i) {
    int d = tid + i * 128;
    float x = h[(long)t * DMODEL + d] + res[(long)t * DMODEL + d];
    v[i] = x; sum += x; sq += x * x;
  }
  blk_reduce2(sum, sq, s1, s2, tid);
  float mu = sum / DMODEL, rs = rsqrtf(sq / DMODEL - mu * mu + 1e-5f);
  sum = 0.f; sq = 0.f;
#pragma unroll
  for (int i = 0; i < 3; ++i) {
    int d = tid + i * 128;
    v[i] = (v[i] - mu) * rs * nfw[d] + nfb[d];
    sum += v[i]; sq += v[i] * v[i];
  }
  blk_reduce2(sum, sq, s1, s2, tid);
  mu = sum / DMODEL; rs = rsqrtf(sq / DMODEL - mu * mu + 1e-5f);
#pragma unroll
  for (int i = 0; i < 3; ++i) {
    int d = tid + i * 128;
    out[(long)t * DMODEL + d] = (v[i] - mu) * rs * n2w[d] + n2b[d];
  }
}
__global__ void write_mask(float* out, const unsigned char* mask) {
  int i = blockIdx.x * blockDim.x + threadIdx.x;
  if (i < GTOT) out[i] = mask[i] ? 1.f : 0.f;
}

// =========================================================
extern "C" void kernel_launch(void* const* d_in, const int* in_sizes, int n_in,
                              void* d_out, int out_size, void* d_ws, size_t ws_size,
                              hipStream_t stream) {
  (void)in_sizes; (void)n_in; (void)out_size; (void)ws_size;

  const float* nb     = (const float*)d_in[0];
  const float* center = (const float*)d_in[1];
  const unsigned char* maskp = (const unsigned char*)d_in[2];
  const float* c1w = (const float*)d_in[3];
  const float* c1b = (const float*)d_in[4];
  const float* bng = (const float*)d_in[5];
  const float* bnb = (const float*)d_in[6];
  const float* bnm = (const float*)d_in[7];
  const float* bnv = (const float*)d_in[8];
  const float* c2w = (const float*)d_in[9];
  const float* c2b = (const float*)d_in[10];
  const float* pw1 = (const float*)d_in[11];
  const float* pb1 = (const float*)d_in[12];
  const float* pw2 = (const float*)d_in[13];
  const float* pb2 = (const float*)d_in[14];
  const float* ipw = (const float*)d_in[15];
  const float* cw  = (const float*)d_in[16];
  const float* cbp = (const float*)d_in[17];
  const float* xpw = (const float*)d_in[18];
  const float* dpw = (const float*)d_in[19];
  const float* dpb = (const float*)d_in[20];
  const float* alog = (const float*)d_in[21];
  const float* dpar = (const float*)d_in[22];
  const float* opw = (const float*)d_in[23];
  const float* lnw = (const float*)d_in[24];
  const float* lnb = (const float*)d_in[25];
  const float* nfw = (const float*)d_in[26];
  const float* nfb = (const float*)d_in[27];
  const float* n2w = (const float*)d_in[28];
  const float* n2b = (const float*)d_in[29];
  float* outp = (float*)d_out;

  // ---- workspace carve-out (all buffers 256B aligned) ----
  char* wp = (char*)d_ws;
  auto alloc = [&](size_t bytes) -> void* {
    void* p = (void*)wp;
    wp += (bytes + 255) & ~(size_t)255;
    return p;
  };
  bf16_t* W1p   = (bf16_t*)alloc((size_t)NEN1 * KENC * 2);
  bf16_t* W2p   = (bf16_t*)alloc((size_t)ENCD * KENC * 2);
  float*  bns   = (float*) alloc(NEN1 * 4);
  float*  bnt   = (float*) alloc(NEN1 * 4);
  float*  gmax  = (float*) alloc((size_t)GTOT * FEATC * 4);
  bf16_t* Fbuf  = (bf16_t*)alloc((size_t)MCH * KENC * 2);
  bf16_t* Gbuf  = (bf16_t*)alloc((size_t)MCH * KENC * 2);
  unsigned* keys = (unsigned*)alloc((size_t)GTOT * ENCD * 4);
  float*  tokens = (float*)alloc((size_t)GTOT * ENCD * 4);
  bf16_t* ipwb  = (bf16_t*)alloc((size_t)DEPTHL * 1536 * 384 * 2);
  bf16_t* xpwb  = (bf16_t*)alloc((size_t)DEPTHL * 64 * 768 * 2);
  bf16_t* dpwb  = (bf16_t*)alloc((size_t)DEPTHL * 768 * 32 * 2);
  bf16_t* opwb  = (bf16_t*)alloc((size_t)DEPTHL * 384 * 768 * 2);
  int*    vis   = (int*)   alloc(NBATCH * LSEQ * 4);
  float*  pmid  = (float*) alloc((size_t)TOK * 128 * 4);
  float*  hbuf  = (float*) alloc((size_t)TOKP * DMODEL * 4);
  float*  resb  = (float*) alloc((size_t)TOKP * DMODEL * 4);
  bf16_t* hn    = (bf16_t*)alloc((size_t)TOKP * DMODEL * 2);
  float*  xzb   = (float*) alloc((size_t)TOKP * 2 * DINNER * 4);
  float*  xcb   = (float*) alloc((size_t)TOKP * DINNER * 4);
  bf16_t* xcbf  = (bf16_t*)alloc((size_t)TOKP * DINNER * 2);
  float*  dblb  = (float*) alloc((size_t)TOKP * 64 * 4);
  bf16_t* dtA   = (bf16_t*)alloc((size_t)TOKP * 32 * 2);
  float*  dtlb  = (float*) alloc((size_t)TOKP * DINNER * 4);
  bf16_t* ymb   = (bf16_t*)alloc((size_t)TOKP * DINNER * 2);

  auto nb256 = [](long n) { return (unsigned)((n + 255) / 256); };
  auto gemm_blocks = [](int Mt, int Nt4) { return (unsigned)((Mt * Nt4 + 7) / 8); };

  // ---- weight prep (bf16 + padding) ----
  pad_convert<<<nb256((long)NEN1 * KENC), 256, 0, stream>>>(W1p, c1w, 1, 546, 546, NEN1, KENC);
  pad_convert<<<nb256((long)ENCD * KENC), 256, 0, stream>>>(W2p, c2w, 1, ENCD, 546, ENCD, KENC);
  pad_convert<<<nb256((long)DEPTHL * 1536 * 384), 256, 0, stream>>>(ipwb, ipw, DEPTHL, 1536, 384, 1536, 384);
  pad_convert<<<nb256((long)DEPTHL * 64 * 768), 256, 0, stream>>>(xpwb, xpw, DEPTHL, 56, 768, 64, 768);
  pad_convert<<<nb256((long)DEPTHL * 768 * 32), 256, 0, stream>>>(dpwb, dpw, DEPTHL, 768, DTR, 768, 32);
  pad_convert<<<nb256((long)DEPTHL * 384 * 768), 256, 0, stream>>>(opwb, opw, DEPTHL, 384, 768, 384, 768);
  bn_prep<<<nb256(NEN1), 256, 0, stream>>>(bns, bnt, bng, bnb, bnm, bnv, c1b);

  // ---- zero-init buffers read in padded regions / accumulated ----
  fill_u32 <<<nb256((long)GTOT * ENCD), 256, 0, stream>>>(keys, 0u, (long)GTOT * ENCD);
  fill_bf16z<<<nb256((long)TOKP * DMODEL), 256, 0, stream>>>(hn, (long)TOKP * DMODEL);
  fill_bf16z<<<nb256((long)TOKP * DINNER), 256, 0, stream>>>(xcbf, (long)TOKP * DINNER);
  fill_bf16z<<<nb256((long)TOKP * DINNER), 256, 0, stream>>>(ymb, (long)TOKP * DINNER);

  // ---- encoder ----
  group_max<<<nb256((long)GTOT * FEATC), 256, 0, stream>>>(gmax, nb);
  for (int ch = 0; ch < GTOT / GCH; ++ch) {
    build_F<<<nb256((long)MCH * KENC), 256, 0, stream>>>(Fbuf, gmax, nb, ch * GCH);
    // conv1 + BN + ReLU -> G (bf16)
    gemm_wmma<1><<<gemm_blocks(MCH / 16, NEN1 / 64), 256, 0, stream>>>(
        Fbuf, W1p, nullptr, Gbuf, bns, bnt, nullptr, 0,
        MCH / 16, NEN1 / 64, KENC / 32, KENC, KENC, KENC);
    // conv2 + max-pool over 32 points (atomic key max)
    gemm_wmma<2><<<gemm_blocks(MCH / 16, ENCD / 64), 256, 0, stream>>>(
        Gbuf, W2p, nullptr, nullptr, nullptr, nullptr, keys, ch * GCH,
        MCH / 16, ENCD / 64, KENC / 32, KENC, KENC, 0);
  }
  finish_tokens<<<nb256((long)GTOT * ENCD), 256, 0, stream>>>(tokens, keys, c2b);

  // ---- visible gather + pos embedding ----
  vis_build<<<1, 32, 0, stream>>>(vis, maskp);
  pos_mid<<<nb256((long)TOK * 128), 256, 0, stream>>>(pmid, center, vis, pw1, pb1);
  pos_h0 <<<nb256((long)TOK * DMODEL), 256, 0, stream>>>(hbuf, tokens, pmid, vis, pw2, pb2);

  // ---- 12 Mamba layers ----
  for (int i = 0; i < DEPTHL; ++i) {
    resln_kernel<<<TOK, 128, 0, stream>>>(hbuf, resb, hn,
                                          lnw + i * DMODEL, lnb + i * DMODEL, i == 0);
    // in_proj: xz = hn @ ipw^T   (824x384x1536)
    gemm_wmma<0><<<gemm_blocks(TOKP / 16, 1536 / 64), 256, 0, stream>>>(
        hn, ipwb + (size_t)i * 1536 * 384, xzb, nullptr, nullptr, nullptr, nullptr, 0,
        TOKP / 16, 1536 / 64, 384 / 32, 384, 384, 1536);
    conv_silu<<<nb256((long)TOK * DINNER), 256, 0, stream>>>(
        xcb, xcbf, xzb, cw + (size_t)i * DINNER * 4, cbp + (size_t)i * DINNER);
    // x_proj: dbl = xc @ xpw^T   (824x768x56->64)
    gemm_wmma<0><<<gemm_blocks(TOKP / 16, 64 / 64), 256, 0, stream>>>(
        xcbf, xpwb + (size_t)i * 64 * 768, dblb, nullptr, nullptr, nullptr, nullptr, 0,
        TOKP / 16, 64 / 64, 768 / 32, 768, 768, 64);
    dt_copy<<<nb256((long)TOKP * 32), 256, 0, stream>>>(dtA, dblb);
    // dt_proj: dtl = dt24 @ dpw^T  (824x24->32x768)
    gemm_wmma<0><<<gemm_blocks(TOKP / 16, 768 / 64), 256, 0, stream>>>(
        dtA, dpwb + (size_t)i * 768 * 32, dtlb, nullptr, nullptr, nullptr, nullptr, 0,
        TOKP / 16, 768 / 64, 1, 32, 32, 768);
    scan_kernel<<<NBATCH * 3, 256, 0, stream>>>(
        dtlb, dpb + (size_t)i * DINNER, xcb, dblb,
        alog + (size_t)i * DINNER * DSTATE, dpar + (size_t)i * DINNER, xzb, ymb);
    // out_proj: h = ymul @ opw^T  (824x768x384)
    gemm_wmma<0><<<gemm_blocks(TOKP / 16, DMODEL / 64), 256, 0, stream>>>(
        ymb, opwb + (size_t)i * 384 * 768, hbuf, nullptr, nullptr, nullptr, nullptr, 0,
        TOKP / 16, DMODEL / 64, 768 / 32, 768, 768, DMODEL);
  }

  // ---- final double LN + outputs ----
  final_ln_kernel<<<TOK, 128, 0, stream>>>(outp, hbuf, resb, nfw, nfb, n2w, n2b);
  write_mask<<<(GTOT + 255) / 256, 256, 0, stream>>>(outp + (size_t)TOK * DMODEL, maskp);
}
